// InfiniteUnderstandingPiMoE_2078764171518
// MI455X (gfx1250) — matrix operand order repaired and checked by reading, not compiled, not run on hardware
//
#include <hip/hip_runtime.h>

#define DEV __device__ __forceinline__

typedef _Float16 h16;
typedef __attribute__((ext_vector_type(16))) _Float16 v16h;
typedef __attribute__((ext_vector_type(8)))  _Float16 v8h;
typedef __attribute__((ext_vector_type(8)))  float    v8f;

constexpr int B = 8, S = 256, D = 1024, U = 1024, E = 4, H = 16, HD = 64;
constexpr int N = B * S;

// ---------------- WMMA fragment helpers (CDNA5 16x16x32 f16 layouts) ----------------
// A (MxK=16x32): lanes 0-15 & 16-31 both hold rows M=lane&15.
//   lo lanes: elems 0..7 -> K=k0+0..7,   elems 8..15 -> K=k0+16..23
//   hi lanes: elems 0..7 -> K=k0+8..15,  elems 8..15 -> K=k0+24..31
DEV v16h load_fragA(const h16* __restrict__ rowptr, int k0, int hi) {
  const v8h* p0 = (const v8h*)(rowptr + k0 + (hi ? 8 : 0));
  const v8h* p1 = (const v8h*)(rowptr + k0 + 16 + (hi ? 8 : 0));
  v8h a = *p0, b = *p1;
  v16h r;
#pragma unroll
  for (int i = 0; i < 8; ++i) { r[i] = a[i]; r[i + 8] = b[i]; }
  return r;
}
// B (KxN=32x16): lane n holds column n=lane&15; same K order as A (lo: 0..7,16..23; hi: 8..15,24..31).
DEV v16h load_fragB(const h16* __restrict__ rowptr, int k0, int hi) {
  const v8h* p0 = (const v8h*)(rowptr + k0 + (hi ? 8 : 0));
  const v8h* p1 = (const v8h*)(rowptr + k0 + 16 + (hi ? 8 : 0));
  v8h a = *p0, b = *p1;
  v16h r;
#pragma unroll
  for (int i = 0; i < 8; ++i) { r[i] = a[i]; r[i + 8] = b[i]; }
  return r;
}
DEV v8f wmma_f16(v16h a, v16h b, v8f c) {
  return __builtin_amdgcn_wmma_f32_16x16x32_f16(false, a, false, b, (short)0, c, false, false);
}
DEV float sigf(float x) { return 1.0f / (1.0f + __expf(-x)); }

// ---------------- register-blocked GEMM ----------------
// C[n][m] = act(sum_k A[n][k]*W[m][k] + b1[m] (+b2[m]))
// act: 0 none, 1 relu, 2 sigmoid, 3 tanh.  Writes fp32 (Cf) and/or f16 (Ch) if non-null.
// Each wave computes a 32x64 tile: 2 row-frags x 4 col-frags = 8 WMMA accumulators,
// so each k-step issues 6 fragment loads for 8 WMMAs (vs 2 loads / 1 WMMA naive).
__global__ void gemm_wmma_kernel(const h16* __restrict__ A, const h16* __restrict__ W,
                                 const float* __restrict__ b1, const float* __restrict__ b2,
                                 float* __restrict__ Cf, h16* __restrict__ Ch,
                                 int K, int M, int act) {
  const int lane = threadIdx.x & 31;
  const int wave = threadIdx.x >> 5;
  const int hi = lane >> 4, lc = lane & 15;
  const int m0 = (blockIdx.x * 4 + wave) * 64;   // 64 output cols per wave
  const int n0 = blockIdx.y * 32;                // 32 output rows per wave
  const h16* arow0 = A + (size_t)(n0 + lc) * K;
  const h16* arow1 = A + (size_t)(n0 + 16 + lc) * K;
  const h16* wrow0 = W + (size_t)(m0 + lc) * K;
  const h16* wrow1 = W + (size_t)(m0 + 16 + lc) * K;
  const h16* wrow2 = W + (size_t)(m0 + 32 + lc) * K;
  const h16* wrow3 = W + (size_t)(m0 + 48 + lc) * K;
  v8f acc[2][4];
#pragma unroll
  for (int i = 0; i < 2; ++i)
#pragma unroll
    for (int j = 0; j < 4; ++j) acc[i][j] = (v8f){};
  for (int k0 = 0; k0 < K; k0 += 32) {
    v16h a0 = load_fragA(arow0, k0, hi);
    v16h a1 = load_fragA(arow1, k0, hi);
    v16h bb0 = load_fragB(wrow0, k0, hi);
    v16h bb1 = load_fragB(wrow1, k0, hi);
    v16h bb2 = load_fragB(wrow2, k0, hi);
    v16h bb3 = load_fragB(wrow3, k0, hi);
    acc[0][0] = wmma_f16(a0, bb0, acc[0][0]);
    acc[0][1] = wmma_f16(a0, bb1, acc[0][1]);
    acc[0][2] = wmma_f16(a0, bb2, acc[0][2]);
    acc[0][3] = wmma_f16(a0, bb3, acc[0][3]);
    acc[1][0] = wmma_f16(a1, bb0, acc[1][0]);
    acc[1][1] = wmma_f16(a1, bb1, acc[1][1]);
    acc[1][2] = wmma_f16(a1, bb2, acc[1][2]);
    acc[1][3] = wmma_f16(a1, bb3, acc[1][3]);
  }
#pragma unroll
  for (int j = 0; j < 4; ++j) {
    const int col = m0 + j * 16 + lc;
    float bv = b1 ? b1[col] : 0.0f;
    if (b2) bv += b2[col];
#pragma unroll
    for (int i = 0; i < 2; ++i) {
#pragma unroll
      for (int r = 0; r < 8; ++r) {
        int row = n0 + i * 16 + r + 8 * hi;    // C layout: lanes16-31 hold rows M=8..15
        float v = acc[i][j][r] + bv;
        if (act == 1) v = fmaxf(v, 0.0f);
        else if (act == 2) v = sigf(v);
        else if (act == 3) v = tanhf(v);
        size_t idx = (size_t)row * M + col;
        if (Cf) Cf[idx] = v;
        if (Ch) Ch[idx] = (h16)v;
      }
    }
  }
}

// ---------------- small utility kernels ----------------
__global__ void zero_f32_kernel(float* p, int n) {
  int i = blockIdx.x * 256 + threadIdx.x; if (i < n) p[i] = 0.0f;
}
__global__ void zero_f16_kernel(h16* p, int n) {
  int i = blockIdx.x * 256 + threadIdx.x; if (i < n) p[i] = (h16)0.0f;
}
__global__ void f32_to_f16_kernel(const float* __restrict__ s, h16* __restrict__ d, int n) {
  int i = blockIdx.x * 256 + threadIdx.x; if (i < n) d[i] = (h16)s[i];
}
__global__ void mul_f16_kernel(const float* __restrict__ a, const float* __restrict__ b,
                               h16* __restrict__ o, int n) {
  int i = blockIdx.x * 256 + threadIdx.x; if (i < n) o[i] = (h16)(a[i] * b[i]);
}

// mean over S for each (b,d)
__global__ void colmean_kernel(const float* __restrict__ x, float* __restrict__ xm) {
  int b = blockIdx.x;
  int d = blockIdx.y * 256 + threadIdx.x;
  float s = 0.0f;
  for (int ss = 0; ss < S; ++ss) s += x[((size_t)b * S + ss) * D + d];
  xm[(size_t)b * D + d] = s * (1.0f / S);
}
// softmax(xm @ rw.T + rb): one wave, lane = b*4+e
__global__ void router_kernel(const float* __restrict__ xm, const float* __restrict__ rw,
                              const float* __restrict__ rb, float* __restrict__ gates) {
  int lane = threadIdx.x & 31;
  int b = lane >> 2, e = lane & 3;
  float s = rb[e];
  for (int d = 0; d < D; ++d) s += xm[(size_t)b * D + d] * rw[(size_t)e * D + d];
  float mx = s;
  for (int off = 1; off < 4; off <<= 1) mx = fmaxf(mx, __shfl_xor(mx, off, 32));
  float ex = __expf(s - mx);
  float sum = ex;
  for (int off = 1; off < 4; off <<= 1) sum += __shfl_xor(sum, off, 32);
  gates[b * E + e] = ex / sum;
}

// LayerNorm (row length L=U), f16 output only (feeds next GEMM)
__global__ void layernorm_kernel(const float* __restrict__ in, const float* __restrict__ g,
                                 const float* __restrict__ bt, h16* __restrict__ oh, int L) {
  __shared__ float ss[8], sq[8];
  int row = blockIdx.x;
  const float* xr = in + (size_t)row * L;
  float s = 0.0f, q = 0.0f;
  for (int i = threadIdx.x; i < L; i += 256) { float v = xr[i]; s += v; q += v * v; }
  for (int off = 16; off >= 1; off >>= 1) { s += __shfl_xor(s, off, 32); q += __shfl_xor(q, off, 32); }
  int w = threadIdx.x >> 5, lane = threadIdx.x & 31;
  if (lane == 0) { ss[w] = s; sq[w] = q; }
  __syncthreads();
  if (w == 0) {
    s = (lane < 8) ? ss[lane] : 0.0f; q = (lane < 8) ? sq[lane] : 0.0f;
    for (int off = 4; off >= 1; off >>= 1) { s += __shfl_xor(s, off, 32); q += __shfl_xor(q, off, 32); }
    if (lane == 0) { ss[0] = s; sq[0] = q; }
  }
  __syncthreads();
  float mean = ss[0] / L;
  float rstd = rsqrtf(sq[0] / L - mean * mean + 1e-5f);
  for (int i = threadIdx.x; i < L; i += 256)
    oh[(size_t)row * L + i] = (h16)((xr[i] - mean) * rstd * g[i] + bt[i]);
}

// decoder LayerNorm + gate-weighted accumulate into output
__global__ void ln_accum_kernel(const float* __restrict__ in, const float* __restrict__ g,
                                const float* __restrict__ bt, const float* __restrict__ gates,
                                int e, float* __restrict__ out) {
  __shared__ float ss[8], sq[8];
  int row = blockIdx.x;
  const float* xr = in + (size_t)row * D;
  float s = 0.0f, q = 0.0f;
  for (int i = threadIdx.x; i < D; i += 256) { float v = xr[i]; s += v; q += v * v; }
  for (int off = 16; off >= 1; off >>= 1) { s += __shfl_xor(s, off, 32); q += __shfl_xor(q, off, 32); }
  int w = threadIdx.x >> 5, lane = threadIdx.x & 31;
  if (lane == 0) { ss[w] = s; sq[w] = q; }
  __syncthreads();
  if (w == 0) {
    s = (lane < 8) ? ss[lane] : 0.0f; q = (lane < 8) ? sq[lane] : 0.0f;
    for (int off = 4; off >= 1; off >>= 1) { s += __shfl_xor(s, off, 32); q += __shfl_xor(q, off, 32); }
    if (lane == 0) { ss[0] = s; sq[0] = q; }
  }
  __syncthreads();
  float mean = ss[0] / D;
  float rstd = rsqrtf(sq[0] / D - mean * mean + 1e-5f);
  float scale = gates[(row / S) * E + e];
  for (int i = threadIdx.x; i < D; i += 256)
    out[(size_t)row * D + i] += scale * ((xr[i] - mean) * rstd * g[i] + bt[i]);
}

// split qkv (fp32, bias already added) into q/k [bh][s][hd] f16 and v^T [bh][hd][s] f16
__global__ void split_qkv_kernel(const float* __restrict__ qkv, h16* __restrict__ qb,
                                 h16* __restrict__ kb, h16* __restrict__ vt) {
  int n = blockIdx.x;
  int b = n / S, s = n % S;
  size_t base = (size_t)n * (3 * U);
  for (int u = threadIdx.x; u < U; u += 256) {
    int h = u >> 6, hd = u & 63;
    size_t bh = (size_t)b * H + h;
    qb[(bh * S + s) * HD + hd] = (h16)qkv[base + u];
    kb[(bh * S + s) * HD + hd] = (h16)qkv[base + U + u];
    vt[(bh * HD + hd) * S + s] = (h16)qkv[base + 2 * U + u];
  }
}

// one wave per (b,h,q-tile of 16): WMMA scores -> softmax -> WMMA P*V
__global__ void attn_kernel(const h16* __restrict__ qb, const h16* __restrict__ kb,
                            const h16* __restrict__ vt, h16* __restrict__ attb) {
  __shared__ h16 P[16 * 256];
  int blk = blockIdx.x;
  int qt = blk & 15;           // S/16 = 16 tiles
  int bh = blk >> 4;           // b*H + h
  int lane = threadIdx.x & 31;
  int hi = lane >> 4, lc = lane & 15;
  int s0 = qt * 16;
  const h16* qrow = qb + ((size_t)bh * S + s0 + lc) * HD;
  v8f acc[16];
#pragma unroll
  for (int t = 0; t < 16; ++t) {
    const h16* krow = kb + ((size_t)bh * S + t * 16 + lc) * HD;
    v8f c = {};
    c = wmma_f16(load_fragA(qrow, 0, hi), load_fragB(krow, 0, hi), c);
    c = wmma_f16(load_fragA(qrow, 32, hi), load_fragB(krow, 32, hi), c);
    acc[t] = c;
  }
  const float scale = 0.125f;  // 1/sqrt(64)
#pragma unroll
  for (int r = 0; r < 8; ++r) {
    float mx = -1e30f;
#pragma unroll
    for (int t = 0; t < 16; ++t) mx = fmaxf(mx, acc[t][r] * scale);
    for (int off = 1; off < 16; off <<= 1) mx = fmaxf(mx, __shfl_xor(mx, off, 32));
    float pv[16], sum = 0.0f;
#pragma unroll
    for (int t = 0; t < 16; ++t) { pv[t] = __expf(acc[t][r] * scale - mx); sum += pv[t]; }
    for (int off = 1; off < 16; off <<= 1) sum += __shfl_xor(sum, off, 32);
    float inv = 1.0f / sum;
    int row = r + 8 * hi;
#pragma unroll
    for (int t = 0; t < 16; ++t) P[row * 256 + t * 16 + lc] = (h16)(pv[t] * inv);
  }
  __syncthreads();
  int b_ = bh >> 4, h_ = bh & 15;
  const h16* prow = P + (size_t)lc * 256;
#pragma unroll
  for (int c4 = 0; c4 < 4; ++c4) {
    const h16* vrow = vt + ((size_t)bh * HD + c4 * 16 + lc) * S;
    v8f c = {};
    for (int ks = 0; ks < 256; ks += 32)
      c = wmma_f16(load_fragA(prow, ks, hi), load_fragB(vrow, ks, hi), c);
#pragma unroll
    for (int r = 0; r < 8; ++r) {
      int row = r + 8 * hi;
      int n = b_ * S + s0 + row;
      int u = h_ * HD + c4 * 16 + lc;
      attb[(size_t)n * U + u] = (h16)c[r];
    }
  }
}

// one LSTM time step: zg = Z[t] + hprev @ whh.T, gate math, write c/h/m.
// hprev/hnext are f16 [16][U] (rows 8..15 zero padding). One wave per 16 hidden cols.
__global__ void lstm_step_kernel(const float* __restrict__ Z, const h16* __restrict__ whh,
                                 const h16* __restrict__ hprev, float* __restrict__ c,
                                 h16* __restrict__ hnext, h16* __restrict__ mb, int t) {
  int lane = threadIdx.x & 31;
  int hi = lane >> 4, lc = lane & 15;
  int u0 = blockIdx.x * 16;
  const h16* arow = hprev + (size_t)lc * U;
  v8f acc[4];
#pragma unroll
  for (int gi = 0; gi < 4; ++gi) acc[gi] = (v8f){};
  for (int k0 = 0; k0 < U; k0 += 32) {
    v16h a = load_fragA(arow, k0, hi);
#pragma unroll
    for (int gi = 0; gi < 4; ++gi) {
      const h16* wrow = whh + (size_t)(gi * U + u0 + lc) * U;
      acc[gi] = wmma_f16(a, load_fragB(wrow, k0, hi), acc[gi]);
    }
  }
  if (hi == 0) {                       // valid batch rows 0..7 live in lanes 0..15
    int u = u0 + lc;
#pragma unroll
    for (int r = 0; r < 8; ++r) {
      int b = r;
      size_t zi = (size_t)(b * S + t) * (4 * U);
      float iv = acc[0][r] + Z[zi + 0 * U + u];
      float fv = acc[1][r] + Z[zi + 1 * U + u];
      float gv = acc[2][r] + Z[zi + 2 * U + u];
      float ov = acc[3][r] + Z[zi + 3 * U + u];
      float cp = c[(size_t)b * U + u];
      float cn = sigf(fv) * cp + sigf(iv) * tanhf(gv);
      float hn = sigf(ov) * tanhf(cn);
      c[(size_t)b * U + u] = cn;
      hnext[(size_t)b * U + u] = (h16)hn;
      mb[(size_t)(b * S + t) * U + u] = (h16)hn;
    }
  }
}

// ---------------- host orchestration ----------------
static inline void gemm(hipStream_t st, const h16* A, const h16* W, const float* b1,
                        const float* b2, float* Cf, h16* Ch, int K, int M, int act) {
  dim3 grid(M / 256, N / 32);   // wave tile 32x64, 4 waves/block along M
  gemm_wmma_kernel<<<grid, 128, 0, st>>>(A, W, b1, b2, Cf, Ch, K, M, act);
}
static inline void cvt(hipStream_t st, const float* s, h16* d, int n) {
  f32_to_f16_kernel<<<(n + 255) / 256, 256, 0, st>>>(s, d, n);
}

extern "C" void kernel_launch(void* const* d_in, const int* in_sizes, int n_in,
                              void* d_out, int out_size, void* d_ws, size_t ws_size,
                              hipStream_t stream) {
  (void)in_sizes; (void)n_in; (void)out_size; (void)ws_size;
  const float* x      = (const float*)d_in[0];
  const float* rw     = (const float*)d_in[1];
  const float* rb     = (const float*)d_in[2];
  const float* enc_w1 = (const float*)d_in[3];
  const float* enc_b1 = (const float*)d_in[4];
  const float* enc_w2 = (const float*)d_in[5];
  const float* enc_b2 = (const float*)d_in[6];
  const float* elg    = (const float*)d_in[7];
  const float* elb    = (const float*)d_in[8];
  const float* wqkv   = (const float*)d_in[9];
  const float* bqkv   = (const float*)d_in[10];
  const float* wo     = (const float*)d_in[11];
  const float* bo     = (const float*)d_in[12];
  const float* wih    = (const float*)d_in[13];
  const float* whh    = (const float*)d_in[14];
  const float* bih    = (const float*)d_in[15];
  const float* bhh    = (const float*)d_in[16];
  const float* uw1    = (const float*)d_in[17];
  const float* ub1    = (const float*)d_in[18];
  const float* uw2    = (const float*)d_in[19];
  const float* ub2    = (const float*)d_in[20];
  const float* ow1    = (const float*)d_in[21];
  const float* ob1    = (const float*)d_in[22];
  const float* ow2    = (const float*)d_in[23];
  const float* ob2    = (const float*)d_in[24];
  const float* dw1    = (const float*)d_in[25];
  const float* db1    = (const float*)d_in[26];
  const float* dw2    = (const float*)d_in[27];
  const float* db2    = (const float*)d_in[28];
  const float* dlg    = (const float*)d_in[29];
  const float* dlb    = (const float*)d_in[30];
  float* out = (float*)d_out;

  // workspace layout (256B aligned bump allocator, lifetime-based reuse)
  char* base = (char*)d_ws;
  size_t off = 0;
  auto alloc = [&](size_t bytes) -> void* {
    void* p = base + off; off += (bytes + 255) & ~(size_t)255; return p;
  };
  float* gates = (float*)alloc((size_t)B * E * 4);
  float* xm    = (float*)alloc((size_t)B * D * 4);
  h16*   xb    = (h16*)alloc((size_t)N * D * 2);
  h16*   wbuf  = (h16*)alloc((size_t)4 * U * U * 2);   // reused per-weight f16 staging
  h16*   whh16 = (h16*)alloc((size_t)4 * U * U * 2);   // persistent per expert
  h16*   hbuf0 = (h16*)alloc((size_t)16 * U * 2);
  h16*   hbuf1 = (h16*)alloc((size_t)16 * U * 2);
  float* cbuf  = (float*)alloc((size_t)8 * U * 4);
  h16*   h1b   = (h16*)alloc((size_t)N * U * 2);       // enc1 out / attb / branch hidden
  h16*   hencb = (h16*)alloc((size_t)N * U * 2);       // enc LN out / att2b
  float* qkvZ  = (float*)alloc((size_t)N * 4 * U * 4); // qkv (N*3U) then Z (N*4U)
  h16*   qb    = (h16*)alloc((size_t)N * U * 2);       // q f16 / later mb
  h16*   kb    = (h16*)alloc((size_t)N * U * 2);       // k f16 / later gtb
  h16*   vt    = (h16*)alloc((size_t)N * U * 2);       // v^T f16
  float* bufA  = (float*)alloc((size_t)N * U * 4);     // pre-LN / sigmoid branch
  float* bufB  = (float*)alloc((size_t)N * U * 4);     // tanh branch
  h16*   mb    = qb;
  h16*   gtb   = kb;
  h16*   attb  = h1b;
  h16*   att2b = hencb;

  // zero output accumulator + routing
  zero_f32_kernel<<<(N * D + 255) / 256, 256, 0, stream>>>(out, N * D);
  colmean_kernel<<<dim3(B, D / 256), 256, 0, stream>>>(x, xm);
  router_kernel<<<1, 32, 0, stream>>>(xm, rw, rb, gates);
  cvt(stream, x, xb, N * D);

  for (int e = 0; e < E; ++e) {
    // encoder: relu(x@w1.T+b1) -> LN(.@w2.T+b2)
    cvt(stream, enc_w1 + (size_t)e * U * D, wbuf, U * D);
    gemm(stream, xb, wbuf, enc_b1 + (size_t)e * U, nullptr, nullptr, h1b, D, U, 1);
    cvt(stream, enc_w2 + (size_t)e * U * U, wbuf, U * U);
    gemm(stream, h1b, wbuf, enc_b2 + (size_t)e * U, nullptr, bufA, nullptr, U, U, 0);
    layernorm_kernel<<<N, 256, 0, stream>>>(bufA, elg + (size_t)e * U, elb + (size_t)e * U,
                                            hencb, U);
    // attention
    cvt(stream, wqkv + (size_t)e * 3 * U * U, wbuf, 3 * U * U);
    gemm(stream, hencb, wbuf, bqkv + (size_t)e * 3 * U, nullptr, qkvZ, nullptr, U, 3 * U, 0);
    split_qkv_kernel<<<N, 256, 0, stream>>>(qkvZ, qb, kb, vt);
    attn_kernel<<<B * H * (S / 16), 32, 0, stream>>>(qb, kb, vt, attb);
    cvt(stream, wo + (size_t)e * U * U, wbuf, U * U);
    gemm(stream, attb, wbuf, bo + (size_t)e * U, nullptr, nullptr, att2b, U, U, 0);
    // LSTM: Z = att2@wih.T + bih + bhh, then 256 sequential steps
    cvt(stream, wih + (size_t)e * 4 * U * U, wbuf, 4 * U * U);
    gemm(stream, att2b, wbuf, bih + (size_t)e * 4 * U, bhh + (size_t)e * 4 * U,
         qkvZ, nullptr, U, 4 * U, 0);
    cvt(stream, whh + (size_t)e * 4 * U * U, whh16, 4 * U * U);
    zero_f16_kernel<<<(16 * U + 255) / 256, 256, 0, stream>>>(hbuf0, 16 * U);
    zero_f16_kernel<<<(16 * U + 255) / 256, 256, 0, stream>>>(hbuf1, 16 * U);
    zero_f32_kernel<<<(8 * U + 255) / 256, 256, 0, stream>>>(cbuf, 8 * U);
    for (int t = 0; t < S; ++t) {
      h16* hp = (t & 1) ? hbuf1 : hbuf0;
      h16* hn = (t & 1) ? hbuf0 : hbuf1;
      lstm_step_kernel<<<U / 16, 32, 0, stream>>>(qkvZ, whh16, hp, cbuf, hn, mb, t);
    }
    // gated branches: sigmoid(relu(m@uw1)@uw2) * tanh(relu(m@ow1)@ow2)
    cvt(stream, uw1 + (size_t)e * U * U, wbuf, U * U);
    gemm(stream, mb, wbuf, ub1 + (size_t)e * U, nullptr, nullptr, h1b, U, U, 1);
    cvt(stream, uw2 + (size_t)e * U * U, wbuf, U * U);
    gemm(stream, h1b, wbuf, ub2 + (size_t)e * U, nullptr, bufA, nullptr, U, U, 2);
    cvt(stream, ow1 + (size_t)e * U * U, wbuf, U * U);
    gemm(stream, mb, wbuf, ob1 + (size_t)e * U, nullptr, nullptr, h1b, U, U, 1);
    cvt(stream, ow2 + (size_t)e * U * U, wbuf, U * U);
    gemm(stream, h1b, wbuf, ob2 + (size_t)e * U, nullptr, bufB, nullptr, U, U, 3);
    mul_f16_kernel<<<(N * U + 255) / 256, 256, 0, stream>>>(bufA, bufB, gtb, N * U);
    // decoder + LN + gated accumulate
    cvt(stream, dw1 + (size_t)e * U * U, wbuf, U * U);
    gemm(stream, gtb, wbuf, db1 + (size_t)e * U, nullptr, nullptr, h1b, U, U, 1);
    cvt(stream, dw2 + (size_t)e * D * U, wbuf, D * U);
    gemm(stream, h1b, wbuf, db2 + (size_t)e * D, nullptr, bufA, nullptr, U, D, 0);
    ln_accum_kernel<<<N, 256, 0, stream>>>(bufA, dlg + (size_t)e * D, dlb + (size_t)e * D,
                                           gates, e, out);
  }
}
